// Model_17188459118643
// MI455X (gfx1250) — matrix operand order repaired and checked by reading, not compiled, not run on hardware
//
#include <hip/hip_runtime.h>

typedef _Float16 h16 __attribute__((ext_vector_type(16)));
typedef float    f8  __attribute__((ext_vector_type(8)));

union H16u { h16 v; uint4 u[2]; _Float16 h[16]; };
union H8u  { uint4 u; _Float16 h[8]; };

__device__ __forceinline__ float fast_rcp(float x) { return __builtin_amdgcn_rcpf(x); }
__device__ __forceinline__ float sigf(float x) { return fast_rcp(1.0f + __expf(-x)); }
#if defined(__gfx1250__) && __has_builtin(__builtin_amdgcn_tanhf)
__device__ __forceinline__ float tanhe(float x) { return __builtin_amdgcn_tanhf(x); }
#else
__device__ __forceinline__ float tanhe(float x) {
  return 2.0f * fast_rcp(1.0f + __expf(-2.0f * x)) - 1.0f;
}
#endif

// ---- CDNA5 split-barrier helpers (workgroup barrier id -1) ----
__device__ __forceinline__ void lds_reads_done_signal() {
  asm volatile("s_wait_dscnt 0x0\n\ts_barrier_signal -1" ::: "memory");
}
__device__ __forceinline__ void barrier_wait_only() {
  asm volatile("s_barrier_wait -1" ::: "memory");
}
__device__ __forceinline__ void lds_writes_done_full_barrier() {
  asm volatile("s_wait_dscnt 0x0\n\ts_barrier_signal -1\n\ts_barrier_wait -1" ::: "memory");
}

#define N_NODES 64
#define HID 128
#define UPD_STRIDE 520   // halfs per batch-row in s_upd (64*8 + 8 pad -> bank rotate)
#define H_STRIDE 136     // halfs per row in s_h (128 + 8 pad -> bank rotate)

// ---------------------------------------------------------------------------
// Kernel 1: neighbor mean, packed to f16 with K padded 6 -> 8 (A-operand ready)
// ---------------------------------------------------------------------------
__global__ void nbr_mean_pack(const float* __restrict__ matrix,
                              const float* __restrict__ feat,
                              _Float16* __restrict__ updG) {
  const int b    = blockIdx.x;
  const int lane = threadIdx.x & 31;
  const int wave = threadIdx.x >> 5;   // 8 waves
  const float* fb = feat + (size_t)b * N_NODES * 6;

  float f0[6], f1[6];
#pragma unroll
  for (int k = 0; k < 6; ++k) {
    f0[k] = fb[lane * 6 + k];
    f1[k] = fb[(lane + 32) * 6 + k];
  }

  for (int i = wave; i < N_NODES; i += 8) {
    const float* mrow = matrix + ((size_t)b * N_NODES + i) * N_NODES;
    float m0 = mrow[lane] > 0.0f ? 1.0f : 0.0f;
    float m1 = mrow[lane + 32] > 0.0f ? 1.0f : 0.0f;
    float cnt = m0 + m1;
    float s[6];
#pragma unroll
    for (int k = 0; k < 6; ++k) s[k] = m0 * f0[k] + m1 * f1[k];
#pragma unroll
    for (int off = 16; off >= 1; off >>= 1) {
      cnt += __shfl_xor(cnt, off, 32);
#pragma unroll
      for (int k = 0; k < 6; ++k) s[k] += __shfl_xor(s[k], off, 32);
    }
    if (lane == 0) {
      float inv = fast_rcp(cnt + 1.0f);
      H8u o;
      o.h[6] = (_Float16)0.0f; o.h[7] = (_Float16)0.0f;
#pragma unroll
      for (int k = 0; k < 6; ++k) o.h[k] = (_Float16)((s[k] + fb[i * 6 + k]) * inv);
      *(uint4*)(updG + ((size_t)b * N_NODES + i) * 8) = o.u;
    }
  }
}

// ---------------------------------------------------------------------------
// Kernel 2: bidirectional LSTM (T=64, H=128) + final FC, WMMA f16 path.
// 16 waves: 0-7 forward, 8-15 backward. Each wave owns 16 hidden units ->
// 4 gate N-tiles; Wh fragments register-resident. Split barriers hide sync
// latency behind the 20-WMMA chain per step.
// ---------------------------------------------------------------------------
__global__ void __launch_bounds__(512)
bilstm_head(const _Float16* __restrict__ updG,
            const float* __restrict__ Wi_f, const float* __restrict__ Wh_f,
            const float* __restrict__ bi_f, const float* __restrict__ bh_f,
            const float* __restrict__ Wi_b, const float* __restrict__ Wh_b,
            const float* __restrict__ bi_b, const float* __restrict__ bh_b,
            const float* __restrict__ fc_w, const float* __restrict__ fc_b,
            const float* __restrict__ device_idx,
            float* __restrict__ out) {
  __shared__ __align__(16) _Float16 s_upd[16 * UPD_STRIDE];
  __shared__ __align__(16) _Float16 s_h[2][16 * H_STRIDE];

  const int tid   = threadIdx.x;
  const int lane  = tid & 31;
  const int wave  = tid >> 5;     // 0..15
  const int dir   = wave >> 3;    // 0 fwd, 1 bwd
  const int w     = wave & 7;     // hidden chunk id
  const int ln15  = lane & 15;
  const int hiH   = lane >> 4;    // lane half selector
  const int hb    = w * 16;
  const int bbase = blockIdx.x * 16;

  // ---- stage upd tile: 16 rows x (64 t x 8 halfs), contiguous per row ----
  {
    const uint4* grow = (const uint4*)(updG + (size_t)(bbase + wave) * (N_NODES * 8));
    uint4* srow = (uint4*)(s_upd + wave * UPD_STRIDE);
    srow[lane]      = grow[lane];
    srow[lane + 32] = grow[lane + 32];
  }
  // ---- zero h state (h_{-1} = 0 for both directions) ----
  {
    unsigned int* hz = (unsigned int*)&s_h[0][0];
    for (int i = tid; i < (2 * 16 * H_STRIDE) / 2; i += 512) hz[i] = 0u;
  }

  const float* Wi = dir ? Wi_b : Wi_f;
  const float* Wh = dir ? Wh_b : Wh_f;
  const float* bi = dir ? bi_b : bi_f;
  const float* bh = dir ? bh_b : bh_f;

  // ---- register-resident B fragments (lane = column, halfs = K) ----
  h16 bWi[4];      // input-projection weights, K 0..5 valid, rest zero
  h16 bWh[16];     // [gate*4 + kchunk]
  float bias[4];   // bi + bh for this lane's column
#pragma unroll
  for (int g = 0; g < 4; ++g) {
    const int col = g * HID + hb + ln15;
    bias[g] = bi[col] + bh[col];
    H16u u;
    u.u[0] = make_uint4(0, 0, 0, 0);
    u.u[1] = make_uint4(0, 0, 0, 0);
    if (hiH == 0) {           // lanes 16..31 carry K=8..15,24..31 -> all zero
#pragma unroll
      for (int k = 0; k < 6; ++k) u.h[k] = (_Float16)Wi[col * 6 + k];
    }
    bWi[g] = u.v;
#pragma unroll
    for (int kc = 0; kc < 4; ++kc) {
      const float4* p = (const float4*)(Wh + (size_t)col * HID + kc * 32 + hiH * 8);
      H16u wv;
#pragma unroll
      for (int q = 0; q < 4; ++q) {
        float4 t4 = p[(q & 1) + (q >> 1) * 4];
        wv.h[4 * q + 0] = (_Float16)t4.x;
        wv.h[4 * q + 1] = (_Float16)t4.y;
        wv.h[4 * q + 2] = (_Float16)t4.z;
        wv.h[4 * q + 3] = (_Float16)t4.w;
      }
      bWh[g * 4 + kc] = wv.v;
    }
  }

  __syncthreads();

  f8 c = {};          // cell state: 16 rows x 16 hidden, C/D layout
  const f8 zf = {};

  for (int it = 0; it < N_NODES; ++it) {
    const int tf = dir ? (N_NODES - 1 - it) : it;

    // ---- read phase: fetch ALL A fragments for this step ----
    H16u ax;
    ax.u[0] = make_uint4(0, 0, 0, 0);
    ax.u[1] = make_uint4(0, 0, 0, 0);
    if (hiH == 0)
      ax.u[0] = *(const uint4*)(s_upd + ln15 * UPD_STRIDE + tf * 8);

    H16u ah[4];
#pragma unroll
    for (int kc = 0; kc < 4; ++kc) {
      const _Float16* hp = &s_h[dir][ln15 * H_STRIDE + kc * 32 + hiH * 8];
      ah[kc].u[0] = *(const uint4*)hp;
      ah[kc].u[1] = *(const uint4*)(hp + 16);
    }

    // reads of h_{t-1} are in flight; once landed, signal so other waves may
    // proceed toward their writes while we crunch WMMAs
    lds_reads_done_signal();

    // ---- compute phase: 4 gate tiles x (1 x-proj + 4 h-chunks) WMMAs ----
    f8 acc[4];
#pragma unroll
    for (int g = 0; g < 4; ++g)
      acc[g] = __builtin_amdgcn_wmma_f32_16x16x32_f16(
          false, ax.v, false, bWi[g], (short)0, zf, false, false);
#pragma unroll
    for (int kc = 0; kc < 4; ++kc) {
#pragma unroll
      for (int g = 0; g < 4; ++g)
        acc[g] = __builtin_amdgcn_wmma_f32_16x16x32_f16(
            false, ah[kc].v, false, bWh[g * 4 + kc], (short)0, acc[g], false, false);
    }

    // all waves must have finished READING h_{t-1} before we overwrite it
    barrier_wait_only();

    // ---- gate math (f32) + cell update + publish h_t as f16 ----
#pragma unroll
    for (int k = 0; k < 8; ++k) {
      float iv = sigf (acc[0][k] + bias[0]);
      float fv = sigf (acc[1][k] + bias[1]);
      float gv = tanhe(acc[2][k] + bias[2]);
      float ov = sigf (acc[3][k] + bias[3]);
      float cv = fv * c[k] + iv * gv;
      c[k] = cv;
      float hv = ov * tanhe(cv);
      s_h[dir][(hiH * 8 + k) * H_STRIDE + hb + ln15] = (_Float16)hv;
    }

    // publish h_t: wait stores, then full signal+wait round
    lds_writes_done_full_barrier();
  }

  // ---- final FC: y = fc_b + fc_w[0]*idx + fc_w[1:129]·h_fwd + fc_w[129:257]·h_bwd
  if (tid < 16) {
    const int r = tid;
    float y = fc_b[0] + fc_w[0] * device_idx[bbase + r];
#pragma unroll 4
    for (int k = 0; k < HID; ++k) y += fc_w[1 + k]       * (float)s_h[0][r * H_STRIDE + k];
#pragma unroll 4
    for (int k = 0; k < HID; ++k) y += fc_w[1 + HID + k] * (float)s_h[1][r * H_STRIDE + k];
    out[bbase + r] = y;
  }
}

// ---------------------------------------------------------------------------
extern "C" void kernel_launch(void* const* d_in, const int* in_sizes, int n_in,
                              void* d_out, int out_size, void* d_ws, size_t ws_size,
                              hipStream_t stream) {
  const float* device_idx = (const float*)d_in[0];
  const float* matrix     = (const float*)d_in[1];
  const float* features   = (const float*)d_in[2];
  const float* Wi_f       = (const float*)d_in[3];
  const float* Wh_f       = (const float*)d_in[4];
  const float* bi_f       = (const float*)d_in[5];
  const float* bh_f       = (const float*)d_in[6];
  const float* Wi_b       = (const float*)d_in[7];
  const float* Wh_b       = (const float*)d_in[8];
  const float* bi_b       = (const float*)d_in[9];
  const float* bh_b       = (const float*)d_in[10];
  const float* fc_w       = (const float*)d_in[11];
  const float* fc_b       = (const float*)d_in[12];
  float* out = (float*)d_out;

  const int B = in_sizes[0];                 // 4096
  _Float16* updG = (_Float16*)d_ws;          // B * 64 * 8 halfs = 4 MB

  nbr_mean_pack<<<B, 256, 0, stream>>>(matrix, features, updG);
  bilstm_head<<<B / 16, 512, 0, stream>>>(updG,
                                          Wi_f, Wh_f, bi_f, bh_f,
                                          Wi_b, Wh_b, bi_b, bh_b,
                                          fc_w, fc_b, device_idx, out);
}
